// DEINA_67714454389311
// MI455X (gfx1250) — matrix operand order repaired and checked by reading, not compiled, not run on hardware
//
#include <hip/hip_runtime.h>
#include <hip/hip_bf16.h>
#include <stdint.h>

typedef __attribute__((ext_vector_type(16))) _Float16 v16h;
typedef __attribute__((ext_vector_type(8)))  _Float16 v8h;
typedef __attribute__((ext_vector_type(8)))  float    v8f;
typedef __attribute__((ext_vector_type(4)))  unsigned int u32x4;
typedef __attribute__((ext_vector_type(8)))  int      i32x8;
typedef __attribute__((ext_vector_type(4)))  int      i32x4;

#define NEG_SLOPE 0.01f
#define B_SZ   1024
#define T_SZ   512
#define D_SZ   16
#define H_SZ   256
#define L_SZ   128
#define LD_SZ  144     // L + D
#define TM1    511     // T - 1
#define LDPAD  160     // 144 padded to multiple of 32

// ---------------- WMMA fragment helpers (wave32, 16x16x32 f16) ----------------

// A-matrix fragment (16x32 f16): lane L holds row M=L%16.
// elements 0..7  -> K = k0 + (L>=16 ? 8 : 0) + e
// elements 8..15 -> K = k0 + 16 + (L>=16 ? 8 : 0) + (e-8)
// 'p' must point at (row, k0 + hi*8); rows are 'ld' halves apart.
__device__ __forceinline__ v16h load_fragA_f16(const _Float16* p) {
  v8h lo = *(const v8h*)(p);
  v8h hh = *(const v8h*)(p + 16);
  v16h f;
#pragma unroll
  for (int i = 0; i < 8; ++i) { f[i] = lo[i]; f[i + 8] = hh[i]; }
  return f;
}

// B-matrix fragment (32x16 f16): lane L holds column N=L%16.
// elements 0..7  -> K = k0 + (L>=16 ? 8 : 0) + e
// elements 8..15 -> K = k0 + 16 + (L>=16 ? 8 : 0) + (e-8)
__device__ __forceinline__ v16h load_fragB_f16(const _Float16* p) {
  v8h lo = *(const v8h*)(p);
  v8h hh = *(const v8h*)(p + 16);
  v16h f;
#pragma unroll
  for (int i = 0; i < 8; ++i) { f[i] = lo[i]; f[i + 8] = hh[i]; }
  return f;
}

// Build A fragment from a 16-float f32 row (K=16 zero-padded to 32),
// with optional affine (gamma,beta) per channel.
__device__ __forceinline__ v16h make_fragA_pad16(const float* row, int dbase,
                                                 const float* gamma, const float* beta) {
  v16h f;
#pragma unroll
  for (int i = 0; i < 8; ++i) {
    float v = row[i];
    if (gamma) v = v * gamma[dbase + i] + beta[dbase + i];
    f[i]     = (_Float16)v;
    f[i + 8] = (_Float16)0.f;
  }
  return f;
}

__device__ __forceinline__ v8f wmma_f16(v16h a, v16h b, v8f c) {
  return __builtin_amdgcn_wmma_f32_16x16x32_f16(false, a, false, b, (short)0, c, false, false);
}

// ---------------- gfx1250 async/tensor data movement ----------------

// LDS aperture keeps the wave-relative LDS byte offset in addr[31:0].
__device__ __forceinline__ uint32_t lds_byte_off(const void* p) {
  return (uint32_t)(uintptr_t)p;
}

// Per-lane 16B global -> LDS async copy (ASYNCcnt).
__device__ __forceinline__ void async_copy_b128(uint32_t lds_off, const void* gptr) {
  asm volatile("global_load_async_to_lds_b128 %0, %1, off"
               :: "v"(lds_off), "v"((uint64_t)(uintptr_t)gptr)
               : "memory");
}
__device__ __forceinline__ void wait_asynccnt0() {
  asm volatile("s_wait_asynccnt 0" ::: "memory");
}

// Tensor Data Mover: 1-D copy of n_dwords (4B units) global -> LDS.
// D# built per CDNA5 ISA 8.3/8.4: group0 {count=1, lds_addr, global_addr, type=2},
// group1 {data_size=2(4B), tensor_dim0=tile_dim0=n_dwords, dim0_stride=n_dwords}.
// This toolchain exposes the 6-arg builtin: (g0, g1, g2, g3, g4, cpol).
__device__ __forceinline__ void tdm_load_1d(uint32_t lds_off, const void* gptr,
                                            uint32_t n_dwords) {
  uint64_t ga = (uint64_t)(uintptr_t)gptr;
  u32x4 g0;
  g0[0] = 1u;                                    // count=1, user mode
  g0[1] = lds_off;                               // lds_addr [63:32]
  g0[2] = (uint32_t)ga;                          // global_addr lo
  g0[3] = (uint32_t)((ga >> 32) & 0x01FFFFFFu) | (2u << 30);  // addr hi | type=2
  i32x8 g1;
  g1[0] = (int)(2u << 16);                       // data_size=2 (4B), mask=0
  g1[1] = (int)((n_dwords & 0xFFFFu) << 16);     // tensor_dim0 lo16 at [63:48]
  g1[2] = (int)((n_dwords >> 16) | (1u << 16));  // tensor_dim0 hi | tensor_dim1=1
  g1[3] = (int)((n_dwords & 0xFFFFu) << 16);     // tile_dim0 at [127:112]
  g1[4] = 0;                                     // tile_dim1=0(unused), tile_dim2=0
  g1[5] = (int)n_dwords;                         // tensor_dim0_stride lo32
  g1[6] = 0;                                     // stride hi, dim1_stride lo
  g1[7] = 0;
  i32x4 z4 = {0, 0, 0, 0};
  i32x8 z8 = {0, 0, 0, 0, 0, 0, 0, 0};
  __builtin_amdgcn_tensor_load_to_lds(g0, g1, z4, z4, z8, 0);
}

// ---------------- Kernel 1: weight conversion f32 -> f16 with K padding ----------------
__global__ void DEINA_prep_kernel(const float* __restrict__ W0, const float* __restrict__ W1,
                                  const float* __restrict__ W2, const float* __restrict__ Bw,
                                  const float* __restrict__ Kw,
                                  _Float16* W0h, _Float16* W1h, _Float16* W2h,
                                  _Float16* Bwh, _Float16* Kwh) {
  int idx = blockIdx.x * blockDim.x + threadIdx.x;
  if (idx < H_SZ * 32) {                       // W0: (256,16) -> (256,32) pad
    int r = idx / 32, k = idx % 32;
    W0h[idx] = (_Float16)(k < D_SZ ? W0[r * D_SZ + k] : 0.f);
    return;
  }
  idx -= H_SZ * 32;
  if (idx < H_SZ * H_SZ) { W1h[idx] = (_Float16)W1[idx]; return; }
  idx -= H_SZ * H_SZ;
  if (idx < L_SZ * H_SZ) { W2h[idx] = (_Float16)W2[idx]; return; }
  idx -= L_SZ * H_SZ;
  if (idx < LD_SZ * 32) {                      // B_w: (144,16) -> (144,32) pad
    int r = idx / 32, k = idx % 32;
    Bwh[idx] = (_Float16)(k < D_SZ ? Bw[r * D_SZ + k] : 0.f);
    return;
  }
  idx -= LD_SZ * 32;
  if (idx < LD_SZ * LDPAD) {                   // K_w: (144,144) -> (144,160) pad
    int r = idx / LDPAD, k = idx % LDPAD;
    Kwh[idx] = (_Float16)(k < LD_SZ ? Kw[r * LD_SZ + k] : 0.f);
  }
}

// ---------------- Kernel 2: scatter xs into y2s x-columns and y0 ----------------
__global__ void DEINA_xcopy_kernel(const float* __restrict__ xs, float* __restrict__ out,
                                   float* __restrict__ y0ws) {
  int idx = blockIdx.x * blockDim.x + threadIdx.x;   // over B*T*D = 8388608
  int b = idx / (T_SZ * D_SZ);
  int rem = idx % (T_SZ * D_SZ);
  int t = rem / D_SZ, d = rem % D_SZ;
  float v = xs[idx];
  if (t >= 1) out[((size_t)b * TM1 + (t - 1)) * LD_SZ + d] = v;
  else        y0ws[b * LD_SZ + d] = v;
}

// ---------------- Kernel 3: fused 3-layer MLP encoder (WMMA) ----------------
#define WAVES_E 2
__global__ __launch_bounds__(64) void DEINA_encoder_kernel(
    const float* __restrict__ xs, const float* __restrict__ gamma, const float* __restrict__ beta,
    const _Float16* __restrict__ W0h, const float* __restrict__ b0,
    const _Float16* __restrict__ W1h, const float* __restrict__ b1,
    const _Float16* __restrict__ W2h, const float* __restrict__ oscale,
    float* __restrict__ out, float* __restrict__ y0ws) {
  __shared__ __align__(16) _Float16 lds_h[WAVES_E][2][16 * H_SZ];  // h0/h1 slabs (32 KB)
  const int lane = threadIdx.x & 31;
  const int wave = threadIdx.x >> 5;
  const int r  = lane & 15;
  const int hi = lane >> 4;
  const int m0 = (blockIdx.x * WAVES_E + wave) * 16;

  _Float16* h0 = lds_h[wave][0];
  _Float16* h1 = lds_h[wave][1];

  // ---- layer 0: x (BN-fused) @ W0^T, K=16 padded to 32 ----
  const float* xrow = xs + (size_t)(m0 + r) * D_SZ + hi * 8;
  v16h a0 = make_fragA_pad16(xrow, hi * 8, gamma, beta);
#pragma unroll 4
  for (int n = 0; n < 16; ++n) {
    v16h bfr = load_fragB_f16(W0h + (size_t)(n * 16 + r) * 32 + hi * 8);
    v8f acc = {};
    acc = wmma_f16(a0, bfr, acc);
    const int col = n * 16 + r;
    const float bias = b0[col];
#pragma unroll
    for (int v = 0; v < 8; ++v) {
      float val = acc[v] + bias;
      val = val >= 0.f ? val : NEG_SLOPE * val;
      h0[(v + 8 * hi) * H_SZ + col] = (_Float16)val;
    }
  }

  // ---- layer 1: h0 @ W1^T, K=256 (8 chunks) ----
#pragma unroll 2
  for (int n = 0; n < 16; ++n) {
    v8f acc = {};
#pragma unroll
    for (int k = 0; k < 8; ++k) {
      v16h a = load_fragA_f16(h0 + r * H_SZ + k * 32 + hi * 8);
      v16h bfr = load_fragB_f16(W1h + (size_t)(n * 16 + r) * H_SZ + k * 32 + hi * 8);
      acc = wmma_f16(a, bfr, acc);
    }
    const int col = n * 16 + r;
    const float bias = b1[col];
#pragma unroll
    for (int v = 0; v < 8; ++v) {
      float val = acc[v] + bias;
      val = val >= 0.f ? val : NEG_SLOPE * val;
      h1[(v + 8 * hi) * H_SZ + col] = (_Float16)val;
    }
  }

  // ---- layer 2: h1 @ W2^T, K=256, *out_scale, scatter ----
#pragma unroll 2
  for (int n = 0; n < 8; ++n) {
    v8f acc = {};
#pragma unroll
    for (int k = 0; k < 8; ++k) {
      v16h a = load_fragA_f16(h1 + r * H_SZ + k * 32 + hi * 8);
      v16h bfr = load_fragB_f16(W2h + (size_t)(n * 16 + r) * H_SZ + k * 32 + hi * 8);
      acc = wmma_f16(a, bfr, acc);
    }
    const int col = n * 16 + r;
    const float sc = oscale[col];
#pragma unroll
    for (int v = 0; v < 8; ++v) {
      const int m = m0 + v + 8 * hi;
      const int b = m >> 9;       // /512
      const int t = m & 511;
      const float g = acc[v] * sc;
      if (t >= 1) out[((size_t)b * TM1 + (t - 1)) * LD_SZ + D_SZ + col] = g;
      else        y0ws[b * LD_SZ + D_SZ + col] = g;
    }
  }
}

// ---------------- Kernel 4: 511-step linear scan y <- y*K^T + u*B^T (WMMA) ----------------
#define SCAN_WAVES 9
__global__ __launch_bounds__(32 * SCAN_WAVES) void DEINA_scan_kernel(
    const float* __restrict__ us, const _Float16* __restrict__ Kwh,
    const _Float16* __restrict__ Bwh, const float* __restrict__ y0ws,
    float* __restrict__ out2) {
  __shared__ __align__(16) _Float16 sK[LD_SZ * LDPAD];   // 45 KB : K_w (144x160)
  __shared__ __align__(16) _Float16 sB[LD_SZ * 32];      //  9 KB : B_w (144x32)
  __shared__ __align__(16) _Float16 sY[16 * LDPAD];      //  5 KB : state tile

  const int tid  = threadIdx.x;
  const int lane = tid & 31;
  const int wave = tid >> 5;
  const int r  = lane & 15;
  const int hi = lane >> 4;
  const int b0 = blockIdx.x * 16;
  const int n0 = wave * 16;
  const int col = n0 + r;

  // K_w: bulk per-lane async global->LDS copies (ASYNCcnt, no VGPR staging)
  {
    const uint32_t sK_off = lds_byte_off(sK);
    const char* gK = (const char*)Kwh;
    for (int i = tid; i < (LD_SZ * LDPAD * 2) / 16; i += 32 * SCAN_WAVES)   // 2880 x 16B
      async_copy_b128(sK_off + i * 16, gK + i * 16);
  }
  // B_w: one Tensor Data Mover descriptor issued by wave 0 (TENSORcnt)
  if (wave == 0) {
    tdm_load_1d(lds_byte_off(sB), Bwh, (LD_SZ * 32 * 2) / 4);               // 2304 dwords
  }
  // y0: f32 -> f16 state tile, zero-pad cols 144..159
  for (int i = tid; i < 16 * LDPAD; i += 32 * SCAN_WAVES) {
    int rr = i / LDPAD, cc = i % LDPAD;
    float v = (cc < LD_SZ) ? y0ws[(b0 + rr) * LD_SZ + cc] : 0.f;
    sY[i] = (_Float16)v;
  }
  wait_asynccnt0();
  if (wave == 0) __builtin_amdgcn_s_wait_tensorcnt(0);
  __syncthreads();

  for (int t = 0; t < TM1; ++t) {
    v8f acc = {};

    // Bu contribution: A from u[:, t, :] (f32 global, K=16 padded), B from B_w
    const float* urow = us + ((size_t)(b0 + r) * TM1 + t) * D_SZ + hi * 8;
    v16h au = make_fragA_pad16(urow, hi * 8, nullptr, nullptr);
    if (t + 1 < TM1) __builtin_prefetch(urow + D_SZ, 0, 1);
    acc = wmma_f16(au, load_fragB_f16(sB + col * 32 + hi * 8), acc);

    // state contribution: 5 K-chunks over the (padded) 160-dim state
#pragma unroll
    for (int k = 0; k < 5; ++k) {
      v16h a = load_fragA_f16(sY + r * LDPAD + k * 32 + hi * 8);
      v16h bfr = load_fragB_f16(sK + col * LDPAD + k * 32 + hi * 8);
      acc = wmma_f16(a, bfr, acc);
    }

    __syncthreads();   // everyone done reading old state

    // write f32 result to y2s_pred and f16 state back to LDS
#pragma unroll
    for (int v = 0; v < 8; ++v) {
      const int row = v + 8 * hi;
      const float val = acc[v];
      out2[(((size_t)(b0 + row)) * TM1 + t) * LD_SZ + col] = val;
      sY[row * LDPAD + col] = (_Float16)val;
    }
    __syncthreads();   // state fully updated before next step's reads
  }
}

// ---------------- Host-side launcher ----------------
extern "C" void kernel_launch(void* const* d_in, const int* in_sizes, int n_in,
                              void* d_out, int out_size, void* d_ws, size_t ws_size,
                              hipStream_t stream) {
  const float* xs     = (const float*)d_in[0];
  const float* us     = (const float*)d_in[1];
  const float* gamma  = (const float*)d_in[2];
  const float* beta   = (const float*)d_in[3];
  const float* W0     = (const float*)d_in[4];
  const float* b0     = (const float*)d_in[5];
  const float* W1     = (const float*)d_in[6];
  const float* b1     = (const float*)d_in[7];
  const float* W2     = (const float*)d_in[8];
  const float* oscale = (const float*)d_in[9];
  const float* Bw     = (const float*)d_in[10];
  const float* Kw     = (const float*)d_in[11];

  float* out_y2s  = (float*)d_out;
  float* out_pred = (float*)d_out + (size_t)B_SZ * TM1 * LD_SZ;

  // workspace layout (f16 weights + f32 y0)
  char* ws = (char*)d_ws;
  size_t o = 0;
  _Float16* W0h = (_Float16*)(ws + o); o += (size_t)H_SZ * 32 * 2;
  _Float16* W1h = (_Float16*)(ws + o); o += (size_t)H_SZ * H_SZ * 2;
  _Float16* W2h = (_Float16*)(ws + o); o += (size_t)L_SZ * H_SZ * 2;
  _Float16* Bwh = (_Float16*)(ws + o); o += (size_t)LD_SZ * 32 * 2;
  _Float16* Kwh = (_Float16*)(ws + o); o += (size_t)LD_SZ * LDPAD * 2;
  o = (o + 255) & ~(size_t)255;
  float* y0ws = (float*)(ws + o);      // 1024*144 f32

  {
    int total = H_SZ * 32 + H_SZ * H_SZ + L_SZ * H_SZ + LD_SZ * 32 + LD_SZ * LDPAD; // 134144
    int blocks = (total + 255) / 256;
    DEINA_prep_kernel<<<blocks, 256, 0, stream>>>(W0, W1, W2, Bw, Kw,
                                                  W0h, W1h, W2h, Bwh, Kwh);
  }
  {
    int total = B_SZ * T_SZ * D_SZ;    // 8388608
    DEINA_xcopy_kernel<<<total / 256, 256, 0, stream>>>(xs, out_y2s, y0ws);
  }
  {
    int blocks = (B_SZ * T_SZ) / (16 * WAVES_E);   // 16384
    DEINA_encoder_kernel<<<blocks, 32 * WAVES_E, 0, stream>>>(
        xs, gamma, beta, W0h, b0, W1h, b1, W2h, oscale, out_y2s, y0ws);
  }
  {
    DEINA_scan_kernel<<<B_SZ / 16, 32 * SCAN_WAVES, 0, stream>>>(
        us, Kwh, Bwh, y0ws, out_pred);
  }
}
